// HierarchicalVoxelAttention_37598143709858
// MI455X (gfx1250) — matrix operand order repaired and checked by reading, not compiled, not run on hardware
//
#include <hip/hip_runtime.h>

// ---------------------------------------------------------------------------
// HierarchicalVoxelAttention for MI455X (gfx1250, wave32, WMMA bf16)
//
// Pipeline:
//   zero -> segsum(atomics) -> macro(divide) ->
//   per branch: qkv GEMM -> flash attention -> out-proj GEMM -> LN ->
//               ff1 GEMM(relu) -> ff2 GEMM -> LN
//   proj0/proj1 small GEMMs against fusion_w column blocks
//   final fused GEMM over h with per-row gathered proj0/proj1 adds
// All GEMMs: f32 in memory, bf16 WMMA (f32 accumulate) on the fly.
// ---------------------------------------------------------------------------

#define D_MODEL 256
#define NPART   200000
#define MS0     4096
#define MS1     256

typedef __attribute__((ext_vector_type(16))) __bf16 bf16x16;
typedef __attribute__((ext_vector_type(8)))  __bf16 bf16x8;
typedef __attribute__((ext_vector_type(8)))  float  f32x8;

// native conversion -> v_cvt_*bf16_f32 (hardware RNE), far fewer VALU than
// the manual bit-trick
static __device__ __forceinline__ __bf16 f2bf(float f) { return (__bf16)f; }

static __device__ __forceinline__ bf16x16 cat8(bf16x8 lo, bf16x8 hi) {
  return __builtin_shufflevector(lo, hi, 0,1,2,3,4,5,6,7,8,9,10,11,12,13,14,15);
}

static __device__ __forceinline__ f32x8 wmma_bf16(bf16x16 a, bf16x16 b, f32x8 c) {
  return __builtin_amdgcn_wmma_f32_16x16x32_bf16(false, a, false, b, (short)0, c,
                                                 false, false);
}

// A-fragment (16x32 bf16) from an LDS tile A_lds[16][32]:
// lane<16: row=lane, K = {k0..k0+15} with k0=0 ; lane>=16: k0=16.
static __device__ __forceinline__ bf16x16 load_a_frag_lds(const __bf16* base,
                                                          int lane) {
  int mrow = lane & 15, half = lane >> 4;
  const __bf16* ap = base + mrow * 32 + half * 16;
  bf16x8 lo = *(const bf16x8*)ap;
  bf16x8 hi = *(const bf16x8*)(ap + 8);
  return cat8(lo, hi);
}

// B-fragment: lane owns column `ncol` of W[N,K] (row-major, stride ldb),
// contiguous K slice of 16 starting at kbase. All offsets are 16B aligned.
static __device__ __forceinline__ bf16x16 load_b_frag_global(
    const float* __restrict__ B, size_t row_off, int kslice) {
  const float4* bp = (const float4*)(B + row_off + kslice);
  float4 b0 = bp[0], b1 = bp[1], b2 = bp[2], b3 = bp[3];
  bf16x16 f;
  f[0]  = f2bf(b0.x); f[1]  = f2bf(b0.y); f[2]  = f2bf(b0.z); f[3]  = f2bf(b0.w);
  f[4]  = f2bf(b1.x); f[5]  = f2bf(b1.y); f[6]  = f2bf(b1.z); f[7]  = f2bf(b1.w);
  f[8]  = f2bf(b2.x); f[9]  = f2bf(b2.y); f[10] = f2bf(b2.z); f[11] = f2bf(b2.w);
  f[12] = f2bf(b3.x); f[13] = f2bf(b3.y); f[14] = f2bf(b3.z); f[15] = f2bf(b3.w);
  return f;
}

// pack two bf16 into one dword for a single 32-bit LDS store
static __device__ __forceinline__ unsigned pack2bf(float a, float b) {
  unsigned short lo = __builtin_bit_cast(unsigned short, f2bf(a));
  unsigned short hi = __builtin_bit_cast(unsigned short, f2bf(b));
  return (unsigned)lo | ((unsigned)hi << 16);
}

// ---------------------------------------------------------------------------
// zero / segment-sum / macro kernels
// ---------------------------------------------------------------------------
__global__ void zero_kernel(float* p, long n) {
  long i = (long)blockIdx.x * 256 + threadIdx.x;
  if (i < n) p[i] = 0.f;
}

__global__ __launch_bounds__(256)
void segsum_kernel(const float* __restrict__ h, const int* __restrict__ seg0,
                   const int* __restrict__ seg1, float* sums0, float* cnt0,
                   float* sums1, float* cnt1) {
  int i = blockIdx.x;
  int d = threadIdx.x;
  int s0 = seg0[i], s1 = seg1[i];
  float v = h[(size_t)i * D_MODEL + d];
  atomicAdd(&sums0[(size_t)s0 * D_MODEL + d], v);
  atomicAdd(&sums1[(size_t)s1 * D_MODEL + d], v);
  if (d == 0) {
    atomicAdd(&cnt0[s0], 1.f);
    atomicAdd(&cnt1[s1], 1.f);
  }
}

__global__ __launch_bounds__(256)
void macro_kernel(const float* __restrict__ sums, const float* __restrict__ cnt,
                  float* __restrict__ macro) {
  int v = blockIdx.x, d = threadIdx.x;
  macro[(size_t)v * D_MODEL + d] =
      sums[(size_t)v * D_MODEL + d] / fmaxf(cnt[v], 1.f);
}

// ---------------------------------------------------------------------------
// Generic GEMM: C[M,N] = act(A[M,K] @ W[N,K]^T + bias), f32 io, bf16 WMMA.
// grid = (M/16, N/128), block = 256 (8 waves; wave w -> N column tile w).
// ---------------------------------------------------------------------------
__global__ __launch_bounds__(256)
void gemm_bf16_kernel(const float* __restrict__ A, int lda,
                      const float* __restrict__ B, int ldb,
                      const float* __restrict__ bias,
                      float* __restrict__ C, int ldc, int K, int relu) {
  __shared__ __align__(16) __bf16 A_lds[16][32];
  int tid = threadIdx.x;
  int lane = tid & 31, wave = tid >> 5;
  int mbase = blockIdx.x * 16;
  int nbase = blockIdx.y * 128 + wave * 16;
  int mrow = lane & 15, half = lane >> 4;
  int ncol = nbase + mrow;                       // B column owned by this lane
  size_t brow = (size_t)ncol * ldb;
  // per-thread A staging slot: thread handles 2 consecutive elements
  int ar = (tid * 2) >> 5, ac = (tid * 2) & 31;
  const float* aptr = A + (size_t)(mbase + ar) * lda + ac;

  f32x8 acc;
#pragma unroll
  for (int i = 0; i < 8; ++i) acc[i] = 0.f;

  for (int kb = 0; kb < K; kb += 32) {
    __syncthreads();
    // stage 16x32 A tile, f32 -> packed bf16 (one b64 load + one b32 DS store)
    float2 av = *(const float2*)(aptr + kb);
    *(unsigned*)&A_lds[ar][ac] = pack2bf(av.x, av.y);
    __syncthreads();

    bf16x16 bfrag = load_b_frag_global(B, brow, kb + half * 16);
    bf16x16 afrag = load_a_frag_lds(&A_lds[0][0], lane);
    acc = wmma_bf16(afrag, bfrag, acc);
  }

  float bv = bias ? bias[ncol] : 0.f;
#pragma unroll
  for (int g = 0; g < 8; ++g) {
    int row = mbase + g + half * 8;
    float v = acc[g] + bv;
    if (relu) v = fmaxf(v, 0.f);
    C[(size_t)row * ldc + ncol] = v;
  }
}

// ---------------------------------------------------------------------------
// Fusion GEMM: out = h @ Wh^T + proj0[seg0[row]] + proj1[seg1[row]]
// (proj1 already contains fusion_b). grid = (M/16, N/128), block 256.
// ---------------------------------------------------------------------------
__global__ __launch_bounds__(256)
void gemm_fusion_kernel(const float* __restrict__ A, int lda,
                        const float* __restrict__ B, int ldb,
                        const float* __restrict__ g0,
                        const int* __restrict__ seg0,
                        const float* __restrict__ g1,
                        const int* __restrict__ seg1,
                        float* __restrict__ C, int ldc, int K) {
  __shared__ __align__(16) __bf16 A_lds[16][32];
  int tid = threadIdx.x;
  int lane = tid & 31, wave = tid >> 5;
  int mbase = blockIdx.x * 16;
  int nbase = blockIdx.y * 128 + wave * 16;
  int mrow = lane & 15, half = lane >> 4;
  int ncol = nbase + mrow;
  size_t brow = (size_t)ncol * ldb;
  int ar = (tid * 2) >> 5, ac = (tid * 2) & 31;
  const float* aptr = A + (size_t)(mbase + ar) * lda + ac;

  f32x8 acc;
#pragma unroll
  for (int i = 0; i < 8; ++i) acc[i] = 0.f;

  for (int kb = 0; kb < K; kb += 32) {
    __syncthreads();
    float2 av = *(const float2*)(aptr + kb);
    *(unsigned*)&A_lds[ar][ac] = pack2bf(av.x, av.y);
    __syncthreads();
    bf16x16 bfrag = load_b_frag_global(B, brow, kb + half * 16);
    bf16x16 afrag = load_a_frag_lds(&A_lds[0][0], lane);
    acc = wmma_bf16(afrag, bfrag, acc);
  }

#pragma unroll
  for (int g = 0; g < 8; ++g) {
    int row = mbase + g + half * 8;
    float v = acc[g];
    v += g0[(size_t)seg0[row] * D_MODEL + ncol];   // L2-resident gathers
    v += g1[(size_t)seg1[row] * D_MODEL + ncol];
    C[(size_t)row * ldc + ncol] = v;
  }
}

// ---------------------------------------------------------------------------
// Flash attention over m tokens, H=4 heads, hd=64.
// qkv: [m, 768] f32 (q | k | v). o: [m, 256] f32 (heads concatenated).
// grid = (m/16, 4); block = 32 (one wave).
// ---------------------------------------------------------------------------
__global__ __launch_bounds__(32)
void attn_kernel(const float* __restrict__ qkv, float* __restrict__ o, int m) {
  int lane = threadIdx.x;
  int qb = blockIdx.x * 16;
  int hh = blockIdx.y;
  int mrow = lane & 15, half = lane >> 4;

  __shared__ __align__(16) float  s_tile[16][32];
  __shared__ __align__(16) __bf16 p_tile[16][32];
  __shared__ __align__(16) __bf16 v_lds[64][32];   // [vdim][key] transposed
  __shared__ float alpha_s[16], l_s[16];

  // Q fragments: two K-steps of 32 (hd = 64); 16 contiguous floats per lane
  bf16x16 qf[2];
#pragma unroll
  for (int ks = 0; ks < 2; ++ks) {
    const float* qp =
        qkv + (size_t)(qb + mrow) * 768 + hh * 64 + ks * 32 + half * 16;
    float4 q0 = ((const float4*)qp)[0];
    float4 q1 = ((const float4*)qp)[1];
    float4 q2 = ((const float4*)(qp + 8))[0];
    float4 q3 = ((const float4*)(qp + 8))[1];
    bf16x16 f;
    f[0]  = f2bf(q0.x); f[1]  = f2bf(q0.y); f[2]  = f2bf(q0.z); f[3]  = f2bf(q0.w);
    f[4]  = f2bf(q1.x); f[5]  = f2bf(q1.y); f[6]  = f2bf(q1.z); f[7]  = f2bf(q1.w);
    f[8]  = f2bf(q2.x); f[9]  = f2bf(q2.y); f[10] = f2bf(q2.z); f[11] = f2bf(q2.w);
    f[12] = f2bf(q3.x); f[13] = f2bf(q3.y); f[14] = f2bf(q3.z); f[15] = f2bf(q3.w);
    qf[ks] = f;
  }

  f32x8 oacc[4];
#pragma unroll
  for (int t = 0; t < 4; ++t)
#pragma unroll
    for (int i = 0; i < 8; ++i) oacc[t][i] = 0.f;

  float m_row = -INFINITY, l_row = 0.f;   // valid in lanes 0..15 (row = lane)
  const float scale = 0.125f;             // 1/sqrt(64)

  for (int kb = 0; kb < m; kb += 32) {
    // S tile = Q @ K^T for 32 keys (two 16-column WMMA tiles)
#pragma unroll
    for (int nt = 0; nt < 2; ++nt) {
      f32x8 s;
#pragma unroll
      for (int i = 0; i < 8; ++i) s[i] = 0.f;
#pragma unroll
      for (int ks = 0; ks < 2; ++ks) {
        int key = kb + nt * 16 + mrow;    // lane owns B column = key
        s = wmma_bf16(qf[ks],
                      load_b_frag_global(qkv, (size_t)key * 768 + 256 + hh * 64,
                                         ks * 32 + half * 16),
                      s);
      }
#pragma unroll
      for (int g = 0; g < 8; ++g)
        s_tile[g + half * 8][nt * 16 + mrow] = s[g];
    }
    // stage V block (32 keys x 64 dims), transposed into LDS as bf16.
    // float4 loads: 16 b128 per lane, coalesced across lanes.
#pragma unroll
    for (int idx = lane; idx < 512; idx += 32) {
      int key = idx >> 4;
      int vd = (idx & 15) * 4;
      float4 v = *(const float4*)&qkv[(size_t)(kb + key) * 768 + 512 +
                                      hh * 64 + vd];
      v_lds[vd + 0][key] = f2bf(v.x);
      v_lds[vd + 1][key] = f2bf(v.y);
      v_lds[vd + 2][key] = f2bf(v.z);
      v_lds[vd + 3][key] = f2bf(v.w);
    }
    __syncthreads();

    // online softmax bookkeeping: lane r owns row r
    if (lane < 16) {
      int r = lane;
      float mx = m_row;
      for (int j = 0; j < 32; ++j) mx = fmaxf(mx, s_tile[r][j] * scale);
      float al = __expf(m_row - mx);
      float sum = 0.f;
      for (int j = 0; j < 32; ++j) {
        float p = __expf(s_tile[r][j] * scale - mx);
        p_tile[r][j] = f2bf(p);
        sum += p;
      }
      l_row = l_row * al + sum;
      m_row = mx;
      alpha_s[r] = al;
    }
    __syncthreads();

    // rescale running O, then O += P @ V
#pragma unroll
    for (int g = 0; g < 8; ++g) {
      float a = alpha_s[g + half * 8];
#pragma unroll
      for (int t = 0; t < 4; ++t) oacc[t][g] *= a;
    }
    bf16x16 pfrag = load_a_frag_lds(&p_tile[0][0], lane);
#pragma unroll
    for (int t = 0; t < 4; ++t) {
      const __bf16* vp = &v_lds[t * 16 + mrow][half * 16];
      bf16x8 vlo = *(const bf16x8*)vp;
      bf16x8 vhi = *(const bf16x8*)(vp + 8);
      oacc[t] = wmma_bf16(pfrag, cat8(vlo, vhi), oacc[t]);
    }
    __syncthreads();   // before next iteration overwrites the LDS tiles
  }

  if (lane < 16) l_s[lane] = l_row;
  __syncthreads();
#pragma unroll
  for (int g = 0; g < 8; ++g) {
    int row = qb + g + half * 8;
    float inv = 1.f / l_s[g + half * 8];
#pragma unroll
    for (int t = 0; t < 4; ++t)
      o[(size_t)row * D_MODEL + hh * 64 + t * 16 + mrow] = oacc[t][g] * inv;
  }
}

// ---------------------------------------------------------------------------
// Residual + LayerNorm: out = LN(x + r) * w + b. One block (256 thr) per row.
// ---------------------------------------------------------------------------
__global__ __launch_bounds__(256)
void ln_kernel(const float* __restrict__ x, const float* __restrict__ r,
               const float* __restrict__ w, const float* __restrict__ b,
               float* __restrict__ out) {
  __shared__ float red[256];
  int tid = threadIdx.x;
  size_t row = blockIdx.x;
  float v = x[row * D_MODEL + tid] + r[row * D_MODEL + tid];

  red[tid] = v;
  __syncthreads();
  for (int s = 128; s > 0; s >>= 1) {
    if (tid < s) red[tid] += red[tid + s];
    __syncthreads();
  }
  float mean = red[0] / (float)D_MODEL;
  __syncthreads();

  float c = v - mean;
  red[tid] = c * c;
  __syncthreads();
  for (int s = 128; s > 0; s >>= 1) {
    if (tid < s) red[tid] += red[tid + s];
    __syncthreads();
  }
  float var = red[0] / (float)D_MODEL;

  out[row * D_MODEL + tid] = c * rsqrtf(var + 1e-5f) * w[tid] + b[tid];
}

// ---------------------------------------------------------------------------
// Host orchestration
// ---------------------------------------------------------------------------
extern "C" void kernel_launch(void* const* d_in, const int* in_sizes, int n_in,
                              void* d_out, int out_size, void* d_ws,
                              size_t ws_size, hipStream_t stream) {
  (void)in_sizes; (void)n_in; (void)out_size; (void)ws_size;

  // setup_inputs() dict order: h, seg0, seg1, params0{12}, params1{12},
  // fusion_w, fusion_b
  const float* h     = (const float*)d_in[0];
  const int*   seg0  = (const int*)d_in[1];
  const int*   seg1  = (const int*)d_in[2];
  void* const* P0    = d_in + 3;
  void* const* P1    = d_in + 15;
  const float* fusion_w = (const float*)d_in[27];
  const float* fusion_b = (const float*)d_in[28];
  float* out = (float*)d_out;

  float* ws = (float*)d_ws;
  size_t off = 0;
  auto alloc = [&](size_t n) { float* p = ws + off; off += n; return p; };

  float* sums0 = alloc((size_t)MS0 * D_MODEL);
  float* cnt0  = alloc(MS0);
  float* sums1 = alloc((size_t)MS1 * D_MODEL);
  float* cnt1  = alloc(MS1);
  size_t zero_n = off;                       // accumulators are contiguous
  float* macro0 = alloc((size_t)MS0 * D_MODEL);
  float* macro1 = alloc((size_t)MS1 * D_MODEL);
  float* qkv0   = alloc((size_t)MS0 * 3 * D_MODEL);
  float* qkv1   = alloc((size_t)MS1 * 3 * D_MODEL);
  float* ao0    = alloc((size_t)MS0 * D_MODEL);
  float* ao1    = alloc((size_t)MS1 * D_MODEL);
  float* tmp0   = alloc((size_t)MS0 * D_MODEL);
  float* tmp1   = alloc((size_t)MS1 * D_MODEL);
  float* x1_0   = alloc((size_t)MS0 * D_MODEL);
  float* x1_1   = alloc((size_t)MS1 * D_MODEL);
  float* ff_0   = alloc((size_t)MS0 * 2 * D_MODEL);
  float* ff_1   = alloc((size_t)MS1 * 2 * D_MODEL);
  float* mo0    = alloc((size_t)MS0 * D_MODEL);
  float* mo1    = alloc((size_t)MS1 * D_MODEL);
  float* proj0  = alloc((size_t)MS0 * D_MODEL);
  float* proj1  = alloc((size_t)MS1 * D_MODEL);

  auto gemm = [&](const float* A, int lda, const float* B, int ldb,
                  const float* bias, float* C, int ldc, int M, int N, int K,
                  int relu) {
    gemm_bf16_kernel<<<dim3(M / 16, N / 128), 256, 0, stream>>>(
        A, lda, B, ldb, bias, C, ldc, K, relu);
  };

  // segment-mean pooling
  zero_kernel<<<(unsigned)((zero_n + 255) / 256), 256, 0, stream>>>(ws,
                                                                    (long)zero_n);
  segsum_kernel<<<NPART, 256, 0, stream>>>(h, seg0, seg1, sums0, cnt0, sums1,
                                           cnt1);
  macro_kernel<<<MS0, 256, 0, stream>>>(sums0, cnt0, macro0);
  macro_kernel<<<MS1, 256, 0, stream>>>(sums1, cnt1, macro1);

  auto branch = [&](const float* macro, int m, void* const* P, float* qkv,
                    float* o, float* tmp, float* x1, float* ff, float* mout) {
    const float* in_w  = (const float*)P[0];
    const float* in_b  = (const float*)P[1];
    const float* out_w = (const float*)P[2];
    const float* out_b = (const float*)P[3];
    const float* ln1_w = (const float*)P[4];
    const float* ln1_b = (const float*)P[5];
    const float* ff1_w = (const float*)P[6];
    const float* ff1_b = (const float*)P[7];
    const float* ff2_w = (const float*)P[8];
    const float* ff2_b = (const float*)P[9];
    const float* ln2_w = (const float*)P[10];
    const float* ln2_b = (const float*)P[11];

    gemm(macro, 256, in_w, 256, in_b, qkv, 768, m, 768, 256, 0);
    attn_kernel<<<dim3(m / 16, 4), 32, 0, stream>>>(qkv, o, m);
    gemm(o, 256, out_w, 256, out_b, tmp, 256, m, 256, 256, 0);
    ln_kernel<<<m, 256, 0, stream>>>(tmp, macro, ln1_w, ln1_b, x1);
    gemm(x1, 256, ff1_w, 256, ff1_b, ff, 512, m, 512, 256, 1);
    gemm(ff, 512, ff2_w, 512, ff2_b, tmp, 256, m, 256, 512, 0);
    ln_kernel<<<m, 256, 0, stream>>>(tmp, x1, ln2_w, ln2_b, mout);
  };

  branch(macro0, MS0, P0, qkv0, ao0, tmp0, x1_0, ff_0, mo0);
  branch(macro1, MS1, P1, qkv1, ao1, tmp1, x1_1, ff_1, mo1);

  // project pooled branch outputs through their fusion_w column blocks
  gemm(mo0, 256, fusion_w + 256, 768, nullptr, proj0, 256, MS0, 256, 256, 0);
  gemm(mo1, 256, fusion_w + 512, 768, fusion_b, proj1, 256, MS1, 256, 256, 0);

  // final fused GEMM with gathered per-particle adds
  gemm_fusion_kernel<<<dim3(NPART / 16, 2), 256, 0, stream>>>(
      h, 256, fusion_w, 768, proj0, seg0, proj1, seg1, out, 256, 256);
}
